// State_VQVAE_60730837566064
// MI455X (gfx1250) — compile-verified, matrix operand order checked by reading, not run on hardware
//
#include <hip/hip_runtime.h>
#include <hip/hip_bf16.h>
#include <stdint.h>

typedef unsigned short u16;
typedef __attribute__((ext_vector_type(16))) __bf16 v16bf;
typedef __attribute__((ext_vector_type(8)))  float  v8f;

#define B_TOK   32768
#define G_GRP   8
#define D_DIM   128
#define K_CODE  256
#define H_DIM   1024
#define BG_ROWS (B_TOK * G_GRP)
#define LN_EPS  1e-5f

// ---------------------------------------------------------------------------
// helpers
// ---------------------------------------------------------------------------
__device__ __forceinline__ u16 f2bf(float f) {
  uint32_t u = __float_as_uint(f);
  return (u16)((u + 0x7fffu + ((u >> 16) & 1u)) >> 16);   // RNE
}

// CDNA5 async copy: global -> LDS, 16 bytes per lane, tracked by ASYNCcnt.
// VDST = LDS byte offset (low 32 bits of generic pointer = LDS aperture
// offset per ISA 10.2), VADDR = 64-bit global address.
__device__ __forceinline__ void async_ld_b128(const u16* gptr, u16* lptr) {
  uint32_t loff = (uint32_t)(uintptr_t)lptr;
  uint64_t gaddr = (uint64_t)(uintptr_t)gptr;
  asm volatile("global_load_async_to_lds_b128 %0, %1, off"
               :: "v"(loff), "v"(gaddr)
               : "memory");
}

template<int N>
__device__ __forceinline__ void wait_async() {
#if __has_builtin(__builtin_amdgcn_s_wait_asynccnt)
  __builtin_amdgcn_s_wait_asynccnt((short)N);
#else
  asm volatile("s_wait_asynccnt %0" :: "i"(N) : "memory");
#endif
}

// Load one 16x32 bf16 WMMA fragment (A-layout per CDNA5 ISA 7.12.2).
// Per lane: rows r=lane&15, half=lane>>4; elements are two contiguous 8-elem
// runs at K = half*8 .. +7 and K = 16+half*8 .. +7  -> two b128 LDS loads.
__device__ __forceinline__ v16bf frag_ld(const u16* base, int hl) {
  union { uint4 q[2]; v16bf v; } u;
  u.q[0] = *(const uint4*)(base + hl * 8);
  u.q[1] = *(const uint4*)(base + 16 + hl * 8);
  return u.v;
}

// ---------------------------------------------------------------------------
// fp32 -> bf16 convert (row-major copy)
// ---------------------------------------------------------------------------
__global__ void cvt_f32_bf16(const float* __restrict__ in,
                             u16* __restrict__ out, int n) {
  int i = blockIdx.x * 256 + threadIdx.x;
  if (i < n) out[i] = f2bf(in[i]);
}

// fp32 W[K,N] -> bf16 Wt[Npad,K] (transposed, zero-padded rows N..Npad-1)
__global__ void cvt_transpose_bf16(const float* __restrict__ w,
                                   u16* __restrict__ wt,
                                   int K, int N, int Npad) {
  int t = blockIdx.x * 256 + threadIdx.x;
  if (t < Npad * K) {
    int n = t / K, k = t - n * K;
    wt[t] = (n < N) ? f2bf(w[(size_t)k * N + n]) : (u16)0;
  }
}

// ---------------------------------------------------------------------------
// WMMA bf16 GEMM:  out[M,N] = act( A[M,K] @ W[K,N] + bias[N] )
// W is pre-transposed/padded: Wt[Npad,K].  Block tile 128x128, K-step 32,
// 8 waves (4x2), each wave 32x64 (2x4 frags).  Double-buffered LDS staged
// with global_load_async_to_lds_b128 (ASYNCcnt pipelining).
// ---------------------------------------------------------------------------
#define MODE_RELU 1
#define MODE_OUTB 2
#define MODE_OUTF 4

template<int MODE>
__global__ __launch_bounds__(256) void gemm_bf16_wmma(
    const u16* __restrict__ A, const u16* __restrict__ Wt,
    const float* __restrict__ bias,
    u16* __restrict__ outb, float* __restrict__ outf,
    int M, int N, int K)
{
  // [m][k] / [n][k], row stride 40 u16 (80 B, multiple of 16 B)
  __shared__ __align__(16) u16 As[2][128 * 40];
  __shared__ __align__(16) u16 Bs[2][128 * 40];

  const int tid  = threadIdx.x;
  const int lane = tid & 31;
  const int wv   = tid >> 5;
  const int wm   = wv >> 1;            // 0..3  (M direction)
  const int wn   = wv & 1;             // 0..1  (N direction)
  const int r    = lane & 15;
  const int hl   = lane >> 4;
  const int m0   = blockIdx.y * 128;
  const int n0   = blockIdx.x * 128;

  // per-thread staging coordinates: 512 uint4 per tile, 2 per thread
  const int ar0 = tid >> 2,          ac0 = (tid & 3) * 8;
  const int ar1 = (tid + 256) >> 2,  ac1 = ((tid + 256) & 3) * 8;

  // issues exactly 4 async-load instructions per wave
  auto stage = [&](int buf, int k0) {
    async_ld_b128(A  + (size_t)(m0 + ar0) * K + k0 + ac0, &As[buf][ar0 * 40 + ac0]);
    async_ld_b128(A  + (size_t)(m0 + ar1) * K + k0 + ac1, &As[buf][ar1 * 40 + ac1]);
    async_ld_b128(Wt + (size_t)(n0 + ar0) * K + k0 + ac0, &Bs[buf][ar0 * 40 + ac0]);
    async_ld_b128(Wt + (size_t)(n0 + ar1) * K + k0 + ac1, &Bs[buf][ar1 * 40 + ac1]);
  };

  v8f acc[2][4];
#pragma unroll
  for (int i = 0; i < 2; ++i)
#pragma unroll
    for (int j = 0; j < 4; ++j)
#pragma unroll
      for (int e = 0; e < 8; ++e) acc[i][j][e] = 0.0f;

  const int nk = K >> 5;
  stage(0, 0);

  for (int ki = 0; ki < nk; ++ki) {
    const int buf = ki & 1;
    if (ki + 1 < nk) {
      stage(buf ^ 1, (ki + 1) * 32);
      wait_async<4>();              // previous 4 (tile ki) complete, in order
    } else {
      wait_async<0>();
    }
    __syncthreads();

    v16bf af[2], bfm[4];
#pragma unroll
    for (int i = 0; i < 2; ++i)
      af[i] = frag_ld(&As[buf][(wm * 32 + i * 16 + r) * 40], hl);
#pragma unroll
    for (int j = 0; j < 4; ++j)
      bfm[j] = frag_ld(&Bs[buf][(wn * 64 + j * 16 + r) * 40], hl);

#pragma unroll
    for (int i = 0; i < 2; ++i)
#pragma unroll
      for (int j = 0; j < 4; ++j)
        acc[i][j] = __builtin_amdgcn_wmma_f32_16x16x32_bf16(
            false, af[i], false, bfm[j], (short)0, acc[i][j], false, false);
    __syncthreads();   // all waves done with buf before it is re-staged
  }

  // --- epilogue: bias (+relu), store bf16 and/or f32 ---
#pragma unroll
  for (int j = 0; j < 4; ++j) {
    int col = n0 + wn * 64 + j * 16 + r;
    if (col < N) {
      float bv = bias[col];
#pragma unroll
      for (int i = 0; i < 2; ++i) {
#pragma unroll
        for (int g = 0; g < 8; ++g) {
          int row = m0 + wm * 32 + i * 16 + g + 8 * hl;
          float v = acc[i][j][g] + bv;
          if (MODE & MODE_RELU) v = v > 0.0f ? v : 0.0f;
          if (MODE & MODE_OUTB) outb[(size_t)row * N + col] = f2bf(v);
          if (MODE & MODE_OUTF) outf[(size_t)row * N + col] = v;
        }
      }
    }
  }
}

// ---------------------------------------------------------------------------
// LayerNorm + ReLU, fp32 in -> bf16 out, one row (1024) per block
// ---------------------------------------------------------------------------
__global__ __launch_bounds__(256) void ln_relu_to_bf16(
    const float* __restrict__ h, const float* __restrict__ g,
    const float* __restrict__ b, u16* __restrict__ out)
{
  __shared__ float red[256];
  const int tid = threadIdx.x;
  const size_t row = blockIdx.x;
  const float4 x = ((const float4*)(h + row * H_DIM))[tid];

  red[tid] = x.x + x.y + x.z + x.w;
  __syncthreads();
  for (int st = 128; st > 0; st >>= 1) {
    if (tid < st) red[tid] += red[tid + st];
    __syncthreads();
  }
  const float mu = red[0] * (1.0f / H_DIM);
  __syncthreads();

  const float d0 = x.x - mu, d1 = x.y - mu, d2 = x.z - mu, d3 = x.w - mu;
  red[tid] = d0 * d0 + d1 * d1 + d2 * d2 + d3 * d3;
  __syncthreads();
  for (int st = 128; st > 0; st >>= 1) {
    if (tid < st) red[tid] += red[tid + st];
    __syncthreads();
  }
  const float rs = rsqrtf(red[0] * (1.0f / H_DIM) + LN_EPS);

  const float4 gg = ((const float4*)g)[tid];
  const float4 bb = ((const float4*)b)[tid];
  float y0 = d0 * rs * gg.x + bb.x;
  float y1 = d1 * rs * gg.y + bb.y;
  float y2 = d2 * rs * gg.z + bb.z;
  float y3 = d3 * rs * gg.w + bb.w;
  y0 = y0 > 0.0f ? y0 : 0.0f;
  y1 = y1 > 0.0f ? y1 : 0.0f;
  y2 = y2 > 0.0f ? y2 : 0.0f;
  y3 = y3 > 0.0f ? y3 : 0.0f;
  uint2 o;
  o.x = (uint32_t)f2bf(y0) | ((uint32_t)f2bf(y1) << 16);
  o.y = (uint32_t)f2bf(y2) | ((uint32_t)f2bf(y3) << 16);
  ((uint2*)(out + row * H_DIM))[tid] = o;
}

// ---------------------------------------------------------------------------
// codebook row norms ||c||^2 (256 rows x 128)
// ---------------------------------------------------------------------------
__global__ void codebook_norms(const float* __restrict__ cb,
                               float* __restrict__ cn) {
  int t = threadIdx.x;
  if (t < K_CODE) {
    const float4* rowp = (const float4*)(cb + (size_t)t * D_DIM);
    float s = 0.0f;
    for (int i = 0; i < 32; ++i) {
      float4 v = rowp[i];
      s += v.x * v.x + v.y * v.y + v.z * v.z + v.w * v.w;
    }
    cn[t] = s;
  }
}

// ---------------------------------------------------------------------------
// VQ argmin via WMMA:  argmin_k ( ||c_k||^2 - 2 * z . c_k )
// Block = 128 z-rows; 8 waves, each wave owns 16 rows x all 256 codes.
// Codebook rows are exactly the column-major B-matrix layout (Bt[n][k]).
// z tile + double-buffered 32-code chunks staged with async global->LDS;
// ||c||^2 preloaded into LDS (no global loads in the inner loop).
// ---------------------------------------------------------------------------
__global__ __launch_bounds__(256) void vq_argmin_wmma(
    const u16* __restrict__ zb,     // [BG,128] bf16
    const u16* __restrict__ cbb,    // [256,128] bf16
    const float* __restrict__ cnorm,
    int* __restrict__ idx)
{
  __shared__ __align__(16) u16 zs[128 * 128];        // 32 KB
  __shared__ __align__(16) u16 cbs[2][32 * 128];     // 2 x 8 KB
  __shared__ float cns[256];                         // 1 KB

  const int tid  = threadIdx.x;
  const int lane = tid & 31;
  const int wv   = tid >> 5;
  const int r    = lane & 15;
  const int hl   = lane >> 4;
  const int row0 = blockIdx.x * 128;

  // stage z tile (8 async instructions per wave)
#pragma unroll
  for (int t = 0; t < 8; ++t) {
    int q = tid + t * 256;
    async_ld_b128(zb + (size_t)row0 * 128 + q * 8, zs + q * 8);
  }
  // stage codebook chunk 0 (2 async instructions per wave)
  auto stage_cb = [&](int buf, int ch) {
#pragma unroll
    for (int t = 0; t < 2; ++t) {
      int q = tid + t * 256;                 // 512 uint4 per 32-code chunk
      async_ld_b128(cbb + (size_t)ch * 32 * 128 + q * 8, &cbs[buf][q * 8]);
    }
  };
  stage_cb(0, 0);
  cns[tid] = cnorm[tid];                     // 256 norms -> LDS
  wait_async<0>();
  __syncthreads();

  v16bf af[4];
#pragma unroll
  for (int ks = 0; ks < 4; ++ks)
    af[ks] = frag_ld(zs + (wv * 16 + r) * 128 + ks * 32, hl);

  float best[8];
  int   bidx[8];
#pragma unroll
  for (int g = 0; g < 8; ++g) { best[g] = 3.4e38f; bidx[g] = 0; }

  for (int ch = 0; ch < 8; ++ch) {           // 8 chunks of 32 codes
    const int buf = ch & 1;
    if (ch + 1 < 8) {
      stage_cb(buf ^ 1, ch + 1);
      wait_async<2>();                       // chunk ch resident (in order)
    } else {
      wait_async<0>();
    }
    __syncthreads();

#pragma unroll
    for (int nt = 0; nt < 2; ++nt) {
      v8f acc;
#pragma unroll
      for (int e = 0; e < 8; ++e) acc[e] = 0.0f;
#pragma unroll
      for (int ks = 0; ks < 4; ++ks) {
        v16bf bfr = frag_ld(&cbs[buf][(nt * 16 + r) * 128 + ks * 32], hl);
        acc = __builtin_amdgcn_wmma_f32_16x16x32_bf16(
            false, af[ks], false, bfr, (short)0, acc, false, false);
      }
      int code = ch * 32 + nt * 16 + r;
      float cn = cns[code];
#pragma unroll
      for (int g = 0; g < 8; ++g) {
        float d = cn - 2.0f * acc[g];
        if (d < best[g]) { best[g] = d; bidx[g] = code; }
      }
    }
    __syncthreads();                         // done with buf before re-stage
  }

  // reduce across the 16 lanes that share each output row
#pragma unroll
  for (int off = 8; off > 0; off >>= 1) {
#pragma unroll
    for (int g = 0; g < 8; ++g) {
      float ov = __shfl_xor(best[g], off, 32);
      int   oi = __shfl_xor(bidx[g], off, 32);
      if (ov < best[g] || (ov == best[g] && oi < bidx[g])) {
        best[g] = ov; bidx[g] = oi;
      }
    }
  }
  if (r == 0) {
#pragma unroll
    for (int g = 0; g < 8; ++g)
      idx[row0 + wv * 16 + g + 8 * hl] = bidx[g];
  }
}

// ---------------------------------------------------------------------------
// loss + histogram:  ssq += ||cb[idx[row]] - z[row]||^2 ; hist[idx[row]]++
// ---------------------------------------------------------------------------
__global__ __launch_bounds__(256) void vq_loss_hist(
    const float* __restrict__ z, const float* __restrict__ cb,
    const int* __restrict__ idx, float* __restrict__ lossAcc,
    unsigned* __restrict__ hist)
{
  __shared__ float red[256];
  const int tid = threadIdx.x;
  const int row = blockIdx.x * 256 + tid;   // grid sized exactly BG/256
  const int c = idx[row];
  atomicAdd(&hist[c], 1u);
  const float4* zr = (const float4*)(z + (size_t)row * D_DIM);
  const float4* cr = (const float4*)(cb + (size_t)c * D_DIM);
  float s = 0.0f;
  for (int i = 0; i < 32; ++i) {
    float4 a = zr[i], b = cr[i];
    float d0 = a.x - b.x, d1 = a.y - b.y, d2 = a.z - b.z, d3 = a.w - b.w;
    s += d0 * d0 + d1 * d1 + d2 * d2 + d3 * d3;
  }
  red[tid] = s;
  __syncthreads();
  for (int st = 128; st > 0; st >>= 1) {
    if (tid < st) red[tid] += red[tid + st];
    __syncthreads();
  }
  if (tid == 0) atomicAdd(lossAcc, red[0]);
}

// ---------------------------------------------------------------------------
// gather: zq[row] = codebook_bf16[idx[row]]   (uint4 = 8 bf16 per thread)
// ---------------------------------------------------------------------------
__global__ void gather_zq(const u16* __restrict__ cbb,
                          const int* __restrict__ idx,
                          u16* __restrict__ zq)
{
  int t = blockIdx.x * 256 + threadIdx.x;   // BG*16 total
  int row = t >> 4, c = t & 15;
  int code = idx[row];
  ((uint4*)zq)[t] = ((const uint4*)cbb)[code * 16 + c];
}

// ---------------------------------------------------------------------------
// small scalar kernels
// ---------------------------------------------------------------------------
__global__ void init_acc(float* lossAcc, unsigned* hist) {
  int t = threadIdx.x;
  if (t == 0) *lossAcc = 0.0f;
  hist[t] = 0u;
}

__global__ void finalize_scalars(const float* lossAcc, const unsigned* hist,
                                 float* out_loss, float* out_perp) {
  __shared__ float red[256];
  int t = threadIdx.x;
  float e = (float)hist[t] * (1.0f / (float)BG_ROWS);
  red[t] = e * logf(e + 1e-10f);
  __syncthreads();
  for (int st = 128; st > 0; st >>= 1) {
    if (t < st) red[t] += red[t + st];
    __syncthreads();
  }
  if (t == 0) {
    *out_perp = expf(-red[0]);
    *out_loss = 1.25f * lossAcc[0] * (1.0f / ((float)BG_ROWS * (float)D_DIM));
  }
}

__global__ void idx_to_float(const int* __restrict__ idx,
                             float* __restrict__ out, int n) {
  int i = blockIdx.x * 256 + threadIdx.x;
  if (i < n) out[i] = (float)idx[i];
}

// ---------------------------------------------------------------------------
// launcher
// ---------------------------------------------------------------------------
extern "C" void kernel_launch(void* const* d_in, const int* in_sizes, int n_in,
                              void* d_out, int out_size, void* d_ws, size_t ws_size,
                              hipStream_t stream)
{
  (void)in_sizes; (void)n_in; (void)out_size; (void)ws_size;

  const float* x        = (const float*)d_in[0];
  const float* enc_w1   = (const float*)d_in[1];
  const float* enc_b1   = (const float*)d_in[2];
  const float* enc_w2   = (const float*)d_in[3];
  const float* enc_b2   = (const float*)d_in[4];
  const float* ln_g     = (const float*)d_in[5];
  const float* ln_b     = (const float*)d_in[6];
  const float* enc_w3   = (const float*)d_in[7];
  const float* enc_b3   = (const float*)d_in[8];
  const float* enc_w4   = (const float*)d_in[9];
  const float* enc_b4   = (const float*)d_in[10];
  const float* codebook = (const float*)d_in[11];
  const float* dec_w1   = (const float*)d_in[12];
  const float* dec_b1   = (const float*)d_in[13];
  const float* dec_w2   = (const float*)d_in[14];
  const float* dec_b2   = (const float*)d_in[15];
  const float* dec_w3   = (const float*)d_in[16];
  const float* dec_b3   = (const float*)d_in[17];

  char* ws = (char*)d_ws;
  size_t off = 0;
  auto alloc = [&](size_t bytes) -> char* {
    char* p = ws + off;
    off = (off + bytes + 255) & ~(size_t)255;
    return p;
  };

  // bf16 inputs / transposed weights  (Wt[Npad][K])
  u16* xb    = (u16*)alloc((size_t)B_TOK * 64 * 2);
  u16* wt1   = (u16*)alloc((size_t)H_DIM * 64 * 2);       // [1024][64]
  u16* wt2   = (u16*)alloc((size_t)H_DIM * H_DIM * 2);    // [1024][1024]
  u16* wt3   = (u16*)alloc((size_t)H_DIM * H_DIM * 2);
  u16* wt4   = (u16*)alloc((size_t)D_DIM * D_DIM * 2);    // [128][128]
  u16* cbb   = (u16*)alloc((size_t)K_CODE * D_DIM * 2);   // row-major codebook
  u16* dwt1  = (u16*)alloc((size_t)H_DIM * H_DIM * 2);
  u16* dwt2  = (u16*)alloc((size_t)H_DIM * H_DIM * 2);
  u16* dwt3  = (u16*)alloc((size_t)128 * H_DIM * 2);      // [128(pad of 64)][1024]
  u16* actA  = (u16*)alloc((size_t)B_TOK * H_DIM * 2);
  u16* actB  = (u16*)alloc((size_t)B_TOK * H_DIM * 2);
  float* f32buf = (float*)alloc((size_t)B_TOK * H_DIM * 4);  // h2, then z
  int* idxbuf   = (int*)alloc((size_t)BG_ROWS * 4);
  float* cn     = (float*)alloc(256 * 4);
  float* lossAcc = (float*)alloc(4);
  unsigned* hist = (unsigned*)alloc(256 * 4);

  auto cvtT = [&](const float* w, u16* wt, int K, int N, int Npad) {
    int n = Npad * K;
    cvt_transpose_bf16<<<(n + 255) / 256, 256, 0, stream>>>(w, wt, K, N, Npad);
  };
  cvt_f32_bf16<<<(B_TOK * 64 + 255) / 256, 256, 0, stream>>>(x, xb, B_TOK * 64);
  cvt_f32_bf16<<<(K_CODE * D_DIM + 255) / 256, 256, 0, stream>>>(
      codebook, cbb, K_CODE * D_DIM);
  cvtT(enc_w1, wt1, 64, H_DIM, H_DIM);
  cvtT(enc_w2, wt2, H_DIM, H_DIM, H_DIM);
  cvtT(enc_w3, wt3, H_DIM, H_DIM, H_DIM);
  cvtT(enc_w4, wt4, D_DIM, D_DIM, D_DIM);
  cvtT(dec_w1, dwt1, H_DIM, H_DIM, H_DIM);
  cvtT(dec_w2, dwt2, H_DIM, H_DIM, H_DIM);
  cvtT(dec_w3, dwt3, H_DIM, 64, 128);

  codebook_norms<<<1, 256, 0, stream>>>(codebook, cn);

  const dim3 blk(256);

  // encoder
  gemm_bf16_wmma<MODE_RELU | MODE_OUTB>
      <<<dim3(H_DIM / 128, B_TOK / 128), blk, 0, stream>>>(
          xb, wt1, enc_b1, actA, nullptr, B_TOK, H_DIM, 64);
  gemm_bf16_wmma<MODE_OUTF>
      <<<dim3(H_DIM / 128, B_TOK / 128), blk, 0, stream>>>(
          actA, wt2, enc_b2, nullptr, f32buf, B_TOK, H_DIM, H_DIM);
  ln_relu_to_bf16<<<B_TOK, blk, 0, stream>>>(f32buf, ln_g, ln_b, actB);
  gemm_bf16_wmma<MODE_RELU | MODE_OUTB>
      <<<dim3(H_DIM / 128, B_TOK / 128), blk, 0, stream>>>(
          actB, wt3, enc_b3, actA, nullptr, B_TOK, H_DIM, H_DIM);
  // enc4: z = z_pre @ w4 + b4  as [BG,128] @ [128,128]; keep z in f32 + bf16
  gemm_bf16_wmma<MODE_OUTB | MODE_OUTF>
      <<<dim3(1, BG_ROWS / 128), blk, 0, stream>>>(
          actA, wt4, enc_b4, actB, f32buf, BG_ROWS, D_DIM, D_DIM);

  // vector quantize
  vq_argmin_wmma<<<BG_ROWS / 128, blk, 0, stream>>>(actB, cbb, cn, idxbuf);
  init_acc<<<1, 256, 0, stream>>>(lossAcc, hist);
  vq_loss_hist<<<BG_ROWS / 256, blk, 0, stream>>>(f32buf, codebook, idxbuf,
                                                  lossAcc, hist);
  gather_zq<<<BG_ROWS * 16 / 256, blk, 0, stream>>>(cbb, idxbuf, actA);

  // decoder
  float* out_f = (float*)d_out;
  gemm_bf16_wmma<MODE_RELU | MODE_OUTB>
      <<<dim3(H_DIM / 128, B_TOK / 128), blk, 0, stream>>>(
          actA, dwt1, dec_b1, actB, nullptr, B_TOK, H_DIM, H_DIM);
  gemm_bf16_wmma<MODE_RELU | MODE_OUTB>
      <<<dim3(H_DIM / 128, B_TOK / 128), blk, 0, stream>>>(
          actB, dwt2, dec_b2, actA, nullptr, B_TOK, H_DIM, H_DIM);
  gemm_bf16_wmma<MODE_OUTF>
      <<<dim3(1, B_TOK / 128), blk, 0, stream>>>(
          actA, dwt3, dec_b3, nullptr, out_f + 1, B_TOK, 64, H_DIM);

  // scalars + indices:  [0]=loss, [1..B*64]=decoded, [..]=perplexity, idx
  finalize_scalars<<<1, 256, 0, stream>>>(lossAcc, hist, out_f,
                                          out_f + 1 + (size_t)B_TOK * 64);
  idx_to_float<<<BG_ROWS / 256, blk, 0, stream>>>(
      idxbuf, out_f + 2 + (size_t)B_TOK * 64, BG_ROWS);
}
